// Decoder_21208548508049
// MI455X (gfx1250) — matrix operand order repaired and checked by reading, not compiled, not run on hardware
//
#include <hip/hip_runtime.h>

// ---------------------------------------------------------------------------
// LIF decoder: rec_x = (x * (1 - spike_seq)) @ W_rec + b_rec,
// spike_seq from a tau=2 LIF scan over inp = x @ W_lif + b_lif.
// T=32, B=8, N=256, D=512 -> GEMM rows M = 65536, K = N = 512.
// GEMMs run bf16 WMMA (v_wmma_f32_16x16x32_bf16) with f32 accumulate.
// B operand staged to LDS via the Tensor Data Mover when available.
// ---------------------------------------------------------------------------

typedef __attribute__((ext_vector_type(16))) __bf16 v16bf;
typedef __attribute__((ext_vector_type(8)))  float  v8f;
typedef unsigned int u32x4 __attribute__((ext_vector_type(4)));
typedef int          i32x8 __attribute__((ext_vector_type(8)));
typedef int          i32x4 __attribute__((ext_vector_type(4)));

#define GEMM_M_TOTAL 65536       // T*B*N
#define DD           512         // D (= K = output width)
#define ELEMS        33554432ull // T*B*N*D
#define BND          1048576u    // B*N*D (per-t stride)

#if __has_builtin(__builtin_amdgcn_tensor_load_to_lds) && \
    __has_builtin(__builtin_amdgcn_s_wait_tensorcnt)
#define USE_TDM 1
#else
#define USE_TDM 0
#endif

// float -> bf16 RNE (storage as ushort; no __bf16 arithmetic anywhere)
__device__ __forceinline__ unsigned short f2bf(float f) {
    unsigned int u = __float_as_uint(f);
    u += 0x7FFFu + ((u >> 16) & 1u);
    return (unsigned short)(u >> 16);
}

// ---------------------------------------------------------------------------
// Prep: swizzle 512x512 f32 row-major W[k][e] into bf16 WMMA-B tile layout:
// tile (kt,nt) covers K=[kt*32,+32) x N=[nt*16,+16); within a tile lane l
// holds 16 bf16: col = nt*16 + (l&15), K = kt*32 + ((j&8)<<1) + ((l&16)>>1) + (j&7).
// ---------------------------------------------------------------------------
__global__ __launch_bounds__(256) void swizzle_w_bf16(
        const float* __restrict__ W, unsigned short* __restrict__ out) {
    int idx  = blockIdx.x * 256 + threadIdx.x;  // 0..262143
    int tile = idx >> 9;
    int r    = idx & 511;
    int l    = r >> 4;
    int j    = r & 15;
    int kt   = tile >> 5;
    int nt   = tile & 31;
    int k    = kt * 32 + ((j & 8) << 1) + ((l & 16) >> 1) + (j & 7);
    int e    = nt * 16 + (l & 15);
    out[idx] = f2bf(W[k * DD + e]);
}

#if USE_TDM
// 1D TDM copy: 4096 bf16 (8KB, one 32x128 pre-swizzled B chunk) -> LDS.
// D# per CDNA5 ISA: group0 {count=1, lds_addr, global_addr,
// type=2}; group1 {data_size=1(2B), tensor_dim0=tile_dim0=4096, dim1=1}.
// This toolchain exposes the 6-arg builtin (g0, g1, g2, g3, g4, cpol).
__device__ __forceinline__ void tdm_load_b_chunk(const unsigned short* src,
                                                 unsigned ldsByteOff) {
    unsigned long long ga = (unsigned long long)(size_t)src;
    u32x4 g0;
    g0[0] = 1u;                                       // count=1 (valid D#)
    g0[1] = ldsByteOff;                               // lds_addr (bytes)
    g0[2] = (unsigned)(ga & 0xFFFFFFFFu);             // global_addr[31:0]
    g0[3] = (unsigned)((ga >> 32) & 0x01FFFFFFu)      // global_addr[56:32]
          | (2u << 30);                               // type=2 ("image")
    i32x8 g1;
    g1[0] = 0x00010000;                               // data_size=1 -> 2 bytes
    g1[1] = (int)(4096u << 16);                       // tensor_dim0 = 4096
    g1[2] = 0x00010000;                               // tensor_dim1 = 1
    g1[3] = (int)(4096u << 16);                       // tile_dim0 = 4096
    g1[4] = 1;                                        // tile_dim1 = 1
    g1[5] = 4096;                                     // tensor_dim0_stride
    g1[6] = 0;
    g1[7] = 0;
    i32x4 z4 = {0, 0, 0, 0};
    i32x8 z8 = {0, 0, 0, 0, 0, 0, 0, 0};
    __builtin_amdgcn_tensor_load_to_lds(g0, g1, z4, z4, z8, 0);
}
#endif

// ---------------------------------------------------------------------------
// Tiled GEMM + bias: Out[M x 512] = A[M x 512] @ Bsw + bias.
// WG tile 128x128, 8 waves (4M x 2N groups), wave = 2x4 WMMA tiles.
// ---------------------------------------------------------------------------
template<bool A_BF16>
__global__ __launch_bounds__(256) void gemm_bias(
        const void* __restrict__ Aop, const unsigned short* __restrict__ Bsw,
        const float* __restrict__ bias, float* __restrict__ Out) {
    __shared__ __align__(128) unsigned char ldsBuf[16384];
    unsigned short* ldsA = (unsigned short*)ldsBuf;            // offset 0
    unsigned short* ldsB = (unsigned short*)(ldsBuf + 8192);   // offset 8192

    const int tid  = threadIdx.x;
    const int lane = tid & 31;
    const int wave = tid >> 5;
    const int wm   = wave & 3;            // M group: tiles 2*wm, 2*wm+1
    const int wn   = wave >> 2;           // N group: tiles 4*wn .. +3
    const int rowBase = blockIdx.x * 128;
    const int colBase = blockIdx.y * 128;

    v8f acc[2][4];
    #pragma unroll
    for (int i = 0; i < 2; ++i)
        #pragma unroll
        for (int j = 0; j < 4; ++j)
            acc[i][j] = (v8f){0.f,0.f,0.f,0.f,0.f,0.f,0.f,0.f};

    for (int kc = 0; kc < 16; ++kc) {     // K chunks of 32
        // ---- stage B chunk (8 swizzled tiles = 8KB contiguous) ----
        const unsigned short* bsrc =
            Bsw + ((size_t)(kc * 32 + blockIdx.y * 8)) * 512;
#if USE_TDM
        if (wave == 0) tdm_load_b_chunk(bsrc, 8192u);
#else
        {
            const unsigned short* src = bsrc + tid * 32;
            unsigned short* dstB = ldsB + tid * 32;
            *(uint4*)(dstB)     = *(const uint4*)(src);
            *(uint4*)(dstB + 8) = *(const uint4*)(src + 8);
        }
#endif
        // ---- stage A chunk (128 x 32) coalesced, scatter to WMMA-A layout ----
        if (A_BF16) {
            const unsigned short* A = (const unsigned short*)Aop;
            const int k0 = (tid & 3) * 8;                   // k' base
            const int h  = tid & 3;
            const int jo = (h & 2) ? 8 : 0;
            #pragma unroll
            for (int p = 0; p < 2; ++p) {                   // 64 rows/pass
                const int row  = p * 64 + (tid >> 2);
                const int mt   = row >> 4;
                const int ln   = (row & 15) + ((h & 1) ? 16 : 0);
                uint4 v = *(const uint4*)(A + (size_t)(rowBase + row) * DD
                                            + kc * 32 + k0);
                *(uint4*)(ldsA + (mt * 32 + ln) * 16 + jo) = v;
            }
        } else {
            const float* A = (const float*)Aop;
            const int k0 = (tid & 7) * 4;                   // k' base
            const int h  = k0 >> 3;
            const int jo = (k0 & 7) + ((h & 2) ? 8 : 0);
            #pragma unroll
            for (int p = 0; p < 4; ++p) {                   // 32 rows/pass
                const int row  = p * 32 + (tid >> 3);
                const int mt   = row >> 4;
                const int ln   = (row & 15) + ((h & 1) ? 16 : 0);
                float4 f = *(const float4*)(A + (size_t)(rowBase + row) * DD
                                              + kc * 32 + k0);
                ushort4 s;
                s.x = f2bf(f.x); s.y = f2bf(f.y);
                s.z = f2bf(f.z); s.w = f2bf(f.w);
                *(ushort4*)(ldsA + (mt * 32 + ln) * 16 + jo) = s;
            }
        }
#if USE_TDM
        __builtin_amdgcn_s_wait_tensorcnt(0);
#endif
        __syncthreads();

        // ---- WMMA compute: 2x4 tiles per wave ----
        v16bf aT[2], bT[4];
        #pragma unroll
        for (int i = 0; i < 2; ++i)
            aT[i] = *(const v16bf*)(ldsA + ((wm * 2 + i) * 32 + lane) * 16);
        #pragma unroll
        for (int j = 0; j < 4; ++j)
            bT[j] = *(const v16bf*)(ldsB + ((wn * 4 + j) * 32 + lane) * 16);
        #pragma unroll
        for (int i = 0; i < 2; ++i)
            #pragma unroll
            for (int j = 0; j < 4; ++j)
                acc[i][j] = __builtin_amdgcn_wmma_f32_16x16x32_bf16(
                    false, aT[i], false, bT[j], (short)0, acc[i][j],
                    false, false);
        __syncthreads();
    }

    // ---- epilogue: bias + store (C layout: VGPR r -> row r + (lane<16?0:8)) ----
    const int halfOff = (lane & 16) >> 1;
    #pragma unroll
    for (int i = 0; i < 2; ++i) {
        const int mrow = rowBase + (wm * 2 + i) * 16 + halfOff;
        #pragma unroll
        for (int j = 0; j < 4; ++j) {
            const int col = colBase + (wn * 4 + j) * 16 + (lane & 15);
            const float bv = bias[col];
            #pragma unroll
            for (int r = 0; r < 8; ++r)
                Out[(size_t)(mrow + r) * DD + col] = acc[i][j][r] + bv;
        }
    }
}

// ---------------------------------------------------------------------------
// LIF recurrence + gate: mem += (inp - mem)/2; sp = H(mem-1);
// g = sp ? 0 : x; mem = sp ? 0 : mem.  Emits g as bf16.
// ---------------------------------------------------------------------------
__global__ __launch_bounds__(256) void lif_gate(
        const float* __restrict__ inp, const float* __restrict__ x,
        const float* __restrict__ tr, unsigned short* __restrict__ g) {
    const size_t j4   = (size_t)blockIdx.x * 256 + threadIdx.x;
    const size_t base = j4 * 4;
    float4 m = ((const float4*)tr)[j4];
    #pragma unroll 4
    for (int t = 0; t < 32; ++t) {
        const size_t off = (size_t)t * BND + base;
        const float4 cur = *(const float4*)(inp + off);
        const float4 xv  = *(const float4*)(x + off);
        m.x += (cur.x - m.x) * 0.5f;
        m.y += (cur.y - m.y) * 0.5f;
        m.z += (cur.z - m.z) * 0.5f;
        m.w += (cur.w - m.w) * 0.5f;
        const bool s0 = (m.x - 1.0f) > 0.f;
        const bool s1 = (m.y - 1.0f) > 0.f;
        const bool s2 = (m.z - 1.0f) > 0.f;
        const bool s3 = (m.w - 1.0f) > 0.f;
        ushort4 gv;
        gv.x = f2bf(s0 ? 0.f : xv.x);
        gv.y = f2bf(s1 ? 0.f : xv.y);
        gv.z = f2bf(s2 ? 0.f : xv.z);
        gv.w = f2bf(s3 ? 0.f : xv.w);
        m.x = s0 ? 0.f : m.x;
        m.y = s1 ? 0.f : m.y;
        m.z = s2 ? 0.f : m.z;
        m.w = s3 ? 0.f : m.w;
        *(ushort4*)(g + off) = gv;
    }
}

// ---------------------------------------------------------------------------
extern "C" void kernel_launch(void* const* d_in, const int* in_sizes, int n_in,
                              void* d_out, int out_size, void* d_ws, size_t ws_size,
                              hipStream_t stream) {
    (void)in_sizes; (void)n_in; (void)out_size; (void)ws_size;
    const float* x    = (const float*)d_in[0];  // [T,B,N,D]
    const float* tr   = (const float*)d_in[1];  // [B,N,D]
    const float* Wlif = (const float*)d_in[2];  // [D,D]
    const float* blif = (const float*)d_in[3];  // [D]
    const float* Wrec = (const float*)d_in[4];  // [D,D]
    const float* brec = (const float*)d_in[5];  // [D]
    float* out = (float*)d_out;

    char* ws = (char*)d_ws;
    unsigned short* wlif_sw = (unsigned short*)(ws);                 // 512 KB
    unsigned short* wrec_sw = (unsigned short*)(ws + (512u << 10));  // 512 KB
    float*          inp     = (float*)(ws + (1u << 20));             // 134 MB
    unsigned short* g       = (unsigned short*)(ws + (1u << 20) + ELEMS * 4ull);

    swizzle_w_bf16<<<1024, 256, 0, stream>>>(Wlif, wlif_sw);
    swizzle_w_bf16<<<1024, 256, 0, stream>>>(Wrec, wrec_sw);

    gemm_bias<false><<<dim3(GEMM_M_TOTAL / 128, DD / 128), 256, 0, stream>>>(
        (const void*)x, wlif_sw, blif, inp);

    lif_gate<<<BND / 4 / 256, 256, 0, stream>>>(inp, x, tr, g);

    gemm_bias<true><<<dim3(GEMM_M_TOTAL / 128, DD / 128), 256, 0, stream>>>(
        (const void*)g, wrec_sw, brec, out);
}
